// Transformer_9955734192516
// MI455X (gfx1250) — matrix operand order repaired and checked by reading, not compiled, not run on hardware
//
#include <hip/hip_runtime.h>
#include <hip/hip_bf16.h>
#include <math.h>

// ---------------- problem constants ----------------
constexpr int VSZ  = 32000;
constexpr int DMOD = 512;
constexpr int HEADS = 8;
constexpr int LNUM = 6;
constexpr int DFFN = 2048;
constexpr int SLEN = 2048;
constexpr int BATCH = 2;
constexpr int DK = DMOD / HEADS;       // 64
constexpr int NTOK = BATCH * SLEN;     // 4096
constexpr float EPS = 1e-5f;

// ---------------- CDNA5 WMMA types ----------------
typedef __attribute__((ext_vector_type(16))) __bf16 v16bf;
typedef __attribute__((ext_vector_type(8)))  float  v8f;

union Frag16 { v16bf v; __bf16 e[16]; };
union Acc8   { v8f   v; float  e[8];  };

// native fp32 -> bf16 (RNE) conversion; backend picks v_cvt on gfx1250
__device__ __forceinline__ __bf16 f2bf(float f) { return (__bf16)f; }

// ---------------- async global->LDS (CDNA5 ASYNCcnt path) -----------------
#if __has_builtin(__builtin_amdgcn_global_load_async_to_lds_b128) && \
    __has_builtin(__builtin_amdgcn_s_wait_asynccnt)
#define USE_ASYNC_LDS 1
// builtin signature: (v4i addrspace(1)* src, v4i addrspace(3)* dst, Ii off, Ii cpol)
typedef int v4i __attribute__((vector_size(16)));
typedef __attribute__((address_space(1))) v4i glb_v4i;
typedef __attribute__((address_space(3))) v4i lds_v4i;
// flat LDS address low 32 bits == LDS byte address (ISA 10.2 aperture rules)
#define TO_LDS(p) ((lds_v4i*)(unsigned)(uintptr_t)(p))
#define TO_GLB(p) ((glb_v4i*)(uintptr_t)(p))
#endif

// ---------------- weight convert: fp32 [K,N] -> bf16 transposed [N,K] -----
__global__ __launch_bounds__(256) void convw_kernel(const float* __restrict__ W,
                                                    __bf16* __restrict__ Wt,
                                                    int K, int N) {
    size_t total = (size_t)K * N;
    size_t idx = (size_t)blockIdx.x * blockDim.x + threadIdx.x;
    if (idx >= total) return;
    int k = (int)(idx / N);
    int n = (int)(idx % N);
    Wt[(size_t)n * K + k] = f2bf(W[idx]);
}

// ---------------- embedding + sinusoidal positional encoding --------------
__global__ __launch_bounds__(256) void embed_kernel(const int* __restrict__ tgt,
                                                    const float* __restrict__ emb,
                                                    float* __restrict__ x,
                                                    __bf16* __restrict__ xb) {
    size_t idx = (size_t)blockIdx.x * blockDim.x + threadIdx.x;  // NTOK*DMOD
    int t = (int)(idx >> 9);          // token index (DMOD = 512)
    int d = (int)(idx & 511);
    int s = t & (SLEN - 1);           // position within sequence
    int tok = tgt[t];
    int dpair = d & ~1;
    float div = __expf(-9.210340371976184f * (float)dpair / (float)DMOD); // ln(10000)
    float arg = (float)s * div;
    float pe = (d & 1) ? __cosf(arg) : __sinf(arg);
    float val = emb[(size_t)tok * DMOD + d] * 22.627416997969522f + pe; // sqrt(512)
    x[idx]  = val;
    xb[idx] = f2bf(val);
}

// ---------------- tiled bf16-WMMA GEMM ------------------------------------
// C[M,N](f32, optional) / Cb[M,N](bf16, optional) = A(bf16 [M,K]) * W + bias
// Wt is bf16 [N,K] (pre-transposed).  flags: 1 = ReLU, 2 = add residual R.
constexpr int BM = 64, BN = 128, BK = 32;

__global__ __launch_bounds__(256) void gemm_bf16_kernel(
    const __bf16* __restrict__ A,
    const __bf16* __restrict__ Wt,
    const float* __restrict__ bias,
    const float* __restrict__ R,
    float* __restrict__ C,
    __bf16* __restrict__ Cb,
    int M, int N, int K, int flags)
{
    __shared__ __bf16 As[BM][BK];   // 4 KB
    __shared__ __bf16 Bs[BN][BK];   // 8 KB

    const int tid  = threadIdx.x;
    const int wave = tid >> 5;
    const int lane = tid & 31;
    const int wm = wave >> 2;       // 0..1
    const int wn = wave & 3;        // 0..3
    const int hi  = lane >> 4;      // half-wave
    const int l16 = lane & 15;
    const int m0 = blockIdx.y * BM;
    const int n0 = blockIdx.x * BN;

    // per-thread staging coordinates (pure bf16 copies)
    const int arow = tid >> 2,        acol = (tid & 3) * 8;    // 1 x b128
    const int brow = tid >> 1,        bcol = (tid & 1) * 16;   // 2 x b128

    Acc8 acc[2][2] = {};

    for (int k0 = 0; k0 < K; k0 += BK) {
        const __bf16* agp = A  + (size_t)(m0 + arow) * K + k0 + acol;
        const __bf16* bgp = Wt + (size_t)(n0 + brow) * K + k0 + bcol;
#ifdef USE_ASYNC_LDS
        __builtin_amdgcn_global_load_async_to_lds_b128(TO_GLB(agp),
                                                       TO_LDS(&As[arow][acol]), 0, 0);
        __builtin_amdgcn_global_load_async_to_lds_b128(TO_GLB(bgp),
                                                       TO_LDS(&Bs[brow][bcol]), 0, 0);
        __builtin_amdgcn_global_load_async_to_lds_b128(TO_GLB(bgp + 8),
                                                       TO_LDS(&Bs[brow][bcol + 8]), 0, 0);
        __builtin_amdgcn_s_wait_asynccnt(0);
#else
        #pragma unroll
        for (int i = 0; i < 8; ++i)  As[arow][acol + i] = agp[i];
        #pragma unroll
        for (int i = 0; i < 16; ++i) Bs[brow][bcol + i] = bgp[i];
#endif
        if (k0 + BK < K) {
            __builtin_prefetch(agp + BK, 0, 0);   // global_prefetch_b8
            __builtin_prefetch(bgp + BK, 0, 0);
        }
        __syncthreads();

        // A fragments: lane<16 holds row M=l16 K{0..15}; lane>=16 K{16..31}
        Frag16 fa[2];
        #pragma unroll
        for (int i = 0; i < 2; ++i) {
            int ar = wm * 32 + i * 16 + l16;
            #pragma unroll
            for (int e = 0; e < 8; ++e) {
                fa[i].e[e]     = As[ar][hi * 16 + e];
                fa[i].e[8 + e] = As[ar][hi * 16 + 8 + e];
            }
        }
        // B fragments: lane<16: b[e] = W[K=e][N=l16]; lane>=16: b[e] = W[K=16+e][N=l16]
        Frag16 fb[2];
        #pragma unroll
        for (int j = 0; j < 2; ++j) {
            int br = wn * 32 + j * 16 + l16;
            #pragma unroll
            for (int e = 0; e < 16; ++e)
                fb[j].e[e] = Bs[br][hi * 16 + e];
        }
        #pragma unroll
        for (int i = 0; i < 2; ++i)
            #pragma unroll
            for (int j = 0; j < 2; ++j)
                acc[i][j].v = __builtin_amdgcn_wmma_f32_16x16x32_bf16(
                    false, fa[i].v, false, fb[j].v, (short)0, acc[i][j].v, false, false);
        __syncthreads();
    }

    // epilogue: C element (row = r + 8*hi, col = l16) per vgpr r
    #pragma unroll
    for (int i = 0; i < 2; ++i) {
        #pragma unroll
        for (int j = 0; j < 2; ++j) {
            int colg = n0 + wn * 32 + j * 16 + l16;
            float bia = bias[colg];
            #pragma unroll
            for (int r = 0; r < 8; ++r) {
                int rowg = m0 + wm * 32 + i * 16 + r + 8 * hi;
                float val = acc[i][j].e[r] + bia;
                if (flags & 1) val = fmaxf(val, 0.0f);
                if (flags & 2) val += R[(size_t)rowg * N + colg];
                if (C)  C [(size_t)rowg * N + colg] = val;
                if (Cb) Cb[(size_t)rowg * N + colg] = f2bf(val);
            }
        }
    }
}

// ---------------- flash attention: one wave per (b, h, 16-query tile) -----
__global__ __launch_bounds__(32) void attention_kernel(
    const __bf16* __restrict__ qb, const __bf16* __restrict__ kb,
    const __bf16* __restrict__ vb, const int* __restrict__ tgt,
    __bf16* __restrict__ ao)
{
    __shared__ __bf16 pT[16][32];   // P tile, row-major [query][key]
    __shared__ __bf16 vT[64][32];   // V tile transposed [dim][key]

    const int lane = threadIdx.x;
    const int hi  = lane >> 4;
    const int l16 = lane & 15;
    const int qt = blockIdx.x;       // 0..SLEN/16-1
    const int h  = blockIdx.y;
    const int b  = blockIdx.z;
    const int q0 = qt * 16;
    const size_t tok0 = (size_t)b * SLEN;
    const int hoff = h * DK;
    const float scale = 0.125f;      // 1/sqrt(64)

    // Q fragments (A layout), DK=64 -> 2 fragments of K32
    Frag16 aQ[2];
    {
        const __bf16* qrow = qb + (tok0 + q0 + l16) * DMOD + hoff;
        #pragma unroll
        for (int kk = 0; kk < 2; ++kk) {
            #pragma unroll
            for (int e = 0; e < 8; ++e) {
                aQ[kk].e[e]     = qrow[kk * 32 + hi * 16 + e];
                aQ[kk].e[8 + e] = qrow[kk * 32 + hi * 16 + 8 + e];
            }
        }
    }

    float mrow[8], lsum[8];
    Acc8 o[4];
    #pragma unroll
    for (int r = 0; r < 8; ++r) { mrow[r] = -3.0e38f; lsum[r] = 0.0f; }
    #pragma unroll
    for (int j = 0; j < 4; ++j)
        #pragma unroll
        for (int r = 0; r < 8; ++r) o[j].e[r] = 0.0f;

    const int nsteps = (qt + 2) >> 1;          // causal bound, 32 keys / step
    for (int step = 0; step < nsteps; ++step) {
        const int k0 = step * 32;

        // stage V^T tile: lane handles key = lane
        {
            const __bf16* vrow = vb + (tok0 + k0 + lane) * DMOD + hoff;
            #pragma unroll
            for (int d = 0; d < 64; ++d) vT[d][lane] = vrow[d];
        }

        // scores = Q @ K^T  (two 16-key subtiles, K-dim = DK = 2x32)
        Acc8 s[2];
        #pragma unroll
        for (int ns = 0; ns < 2; ++ns) {
            #pragma unroll
            for (int r = 0; r < 8; ++r) s[ns].e[r] = 0.0f;
            const __bf16* krow = kb + (tok0 + k0 + ns * 16 + l16) * DMOD + hoff;
            #pragma unroll
            for (int kk = 0; kk < 2; ++kk) {
                Frag16 bK;
                #pragma unroll
                for (int e = 0; e < 16; ++e)
                    bK.e[e] = krow[kk * 32 + hi * 16 + e];
                s[ns].v = __builtin_amdgcn_wmma_f32_16x16x32_bf16(
                    false, aQ[kk].v, false, bK.v, (short)0, s[ns].v, false, false);
            }
        }

        // mask + online softmax
        const int key0 = k0 + l16;
        const int key1 = k0 + 16 + l16;
        const bool pad0 = tgt[tok0 + key0] != 0;
        const bool pad1 = tgt[tok0 + key1] != 0;
        float tmax[8];
        #pragma unroll
        for (int r = 0; r < 8; ++r) {
            int qrow_ = q0 + r + 8 * hi;
            float s0 = (pad0 && key0 <= qrow_) ? s[0].e[r] * scale : -1e9f;
            float s1 = (pad1 && key1 <= qrow_) ? s[1].e[r] * scale : -1e9f;
            s[0].e[r] = s0; s[1].e[r] = s1;
            float tm = fmaxf(s0, s1);
            tm = fmaxf(tm, __shfl_xor(tm, 1, 32));
            tm = fmaxf(tm, __shfl_xor(tm, 2, 32));
            tm = fmaxf(tm, __shfl_xor(tm, 4, 32));
            tm = fmaxf(tm, __shfl_xor(tm, 8, 32));
            tmax[r] = tm;
        }
        #pragma unroll
        for (int r = 0; r < 8; ++r) {
            float mnew = fmaxf(mrow[r], tmax[r]);
            float corr = __expf(mrow[r] - mnew);
            float p0 = __expf(s[0].e[r] - mnew);
            float p1 = __expf(s[1].e[r] - mnew);
            float ps = p0 + p1;
            ps += __shfl_xor(ps, 1, 32);
            ps += __shfl_xor(ps, 2, 32);
            ps += __shfl_xor(ps, 4, 32);
            ps += __shfl_xor(ps, 8, 32);
            lsum[r] = lsum[r] * corr + ps;
            mrow[r] = mnew;
            #pragma unroll
            for (int j = 0; j < 4; ++j) o[j].e[r] *= corr;
            pT[r + 8 * hi][l16]      = f2bf(p0);
            pT[r + 8 * hi][16 + l16] = f2bf(p1);
        }
        __syncthreads();

        // O += P @ V  (P is A-matrix 16x32; V^T rows give B fragments)
        Frag16 aP;
        #pragma unroll
        for (int e = 0; e < 8; ++e) {
            aP.e[e]     = pT[l16][hi * 16 + e];
            aP.e[8 + e] = pT[l16][hi * 16 + 8 + e];
        }
        #pragma unroll
        for (int j = 0; j < 4; ++j) {
            Frag16 bV;
            #pragma unroll
            for (int e = 0; e < 16; ++e)
                bV.e[e] = vT[j * 16 + l16][hi * 16 + e];
            o[j].v = __builtin_amdgcn_wmma_f32_16x16x32_bf16(
                false, aP.v, false, bV.v, (short)0, o[j].v, false, false);
        }
        __syncthreads();
    }

    // write normalized output (bf16, consumed only by O-projection GEMM)
    #pragma unroll
    for (int j = 0; j < 4; ++j)
        #pragma unroll
        for (int r = 0; r < 8; ++r) {
            size_t rowg = tok0 + q0 + r + 8 * hi;
            ao[rowg * DMOD + hoff + j * 16 + l16] = f2bf(o[j].e[r] / lsum[r]);
        }
}

// ---------------- layernorm (per token row, D=512) ------------------------
__global__ __launch_bounds__(256) void layernorm_kernel(
    const float* __restrict__ X, const float* __restrict__ g,
    const float* __restrict__ be, float* __restrict__ Y,
    __bf16* __restrict__ Yb)
{
    __shared__ float r1[256], r2[256];
    const int row = blockIdx.x;
    const float* xr = X + (size_t)row * DMOD;
    float s1 = 0.f, s2 = 0.f;
    for (int i = threadIdx.x; i < DMOD; i += 256) {
        float f = xr[i]; s1 += f; s2 += f * f;
    }
    r1[threadIdx.x] = s1; r2[threadIdx.x] = s2;
    __syncthreads();
    for (int off = 128; off > 0; off >>= 1) {
        if (threadIdx.x < off) {
            r1[threadIdx.x] += r1[threadIdx.x + off];
            r2[threadIdx.x] += r2[threadIdx.x + off];
        }
        __syncthreads();
    }
    float mean = r1[0] * (1.0f / DMOD);
    float var  = r2[0] * (1.0f / DMOD) - mean * mean;
    float inv  = rsqrtf(var + EPS);
    for (int i = threadIdx.x; i < DMOD; i += 256) {
        float val = (xr[i] - mean) * inv * g[i] + be[i];
        Y [(size_t)row * DMOD + i] = val;
        Yb[(size_t)row * DMOD + i] = f2bf(val);
    }
}

// ---------------- host orchestration --------------------------------------
extern "C" void kernel_launch(void* const* d_in, const int* in_sizes, int n_in,
                              void* d_out, int out_size, void* d_ws, size_t ws_size,
                              hipStream_t stream)
{
    const int*   tgt  = (const int*)  d_in[0];
    const float* emb  = (const float*)d_in[1];
    const float* Wq   = (const float*)d_in[2];
    const float* bq   = (const float*)d_in[3];
    const float* Wk   = (const float*)d_in[4];
    const float* bk   = (const float*)d_in[5];
    const float* Wv   = (const float*)d_in[6];
    const float* bv   = (const float*)d_in[7];
    const float* Wo   = (const float*)d_in[8];
    const float* bo   = (const float*)d_in[9];
    const float* g1   = (const float*)d_in[10];
    const float* be1  = (const float*)d_in[11];
    const float* Wf1  = (const float*)d_in[12];
    const float* bf1  = (const float*)d_in[13];
    const float* Wf2  = (const float*)d_in[14];
    const float* bf2  = (const float*)d_in[15];
    const float* g3   = (const float*)d_in[16];
    const float* be3  = (const float*)d_in[17];
    const float* gN   = (const float*)d_in[18];
    const float* beN  = (const float*)d_in[19];
    const float* Wout = (const float*)d_in[20];
    const float* bout = (const float*)d_in[21];
    float* out = (float*)d_out;

    // workspace carve-out
    char* p = (char*)d_ws;
    auto carve = [&](size_t bytes) -> void* {
        void* r = (void*)p;
        p += (bytes + 255) & ~(size_t)255;
        return r;
    };
    const size_t F32SZ = (size_t)NTOK * DMOD * sizeof(float);    // 8 MB
    const size_t B16SZ = (size_t)NTOK * DMOD * sizeof(__bf16);   // 4 MB
    float*  x    = (float*) carve(F32SZ);
    float*  t    = (float*) carve(F32SZ);
    __bf16* xb   = (__bf16*)carve(B16SZ);
    __bf16* tb   = (__bf16*)carve(B16SZ);
    __bf16* q16  = (__bf16*)carve(B16SZ);
    __bf16* k16  = (__bf16*)carve(B16SZ);
    __bf16* v16  = (__bf16*)carve(B16SZ);
    __bf16* ao16 = (__bf16*)carve(B16SZ);
    __bf16* h16  = (__bf16*)carve((size_t)NTOK * DFFN * sizeof(__bf16)); // 16 MB
    __bf16* cwq  = (__bf16*)carve((size_t)DMOD * DMOD * 2);
    __bf16* cwk  = (__bf16*)carve((size_t)DMOD * DMOD * 2);
    __bf16* cwv  = (__bf16*)carve((size_t)DMOD * DMOD * 2);
    __bf16* cwo  = (__bf16*)carve((size_t)DMOD * DMOD * 2);
    __bf16* cw1  = (__bf16*)carve((size_t)DMOD * DFFN * 2);
    __bf16* cw2  = (__bf16*)carve((size_t)DFFN * DMOD * 2);
    __bf16* cwT  = (__bf16*)carve((size_t)DMOD * VSZ * 2);       // 32.8 MB

    auto convw = [&](const float* W, __bf16* Wt, int K, int N) {
        size_t total = (size_t)K * N;
        int blocks = (int)((total + 255) / 256);
        convw_kernel<<<blocks, 256, 0, stream>>>(W, Wt, K, N);
    };
    auto gemm = [&](const __bf16* A, const __bf16* Wt, const float* bias,
                    const float* R, float* C, __bf16* Cb, int N, int K, int flags) {
        dim3 grid(N / BN, NTOK / BM);
        gemm_bf16_kernel<<<grid, 256, 0, stream>>>(A, Wt, bias, R, C, Cb,
                                                   NTOK, N, K, flags);
    };

    // embedding + PE
    embed_kernel<<<(NTOK * DMOD) / 256, 256, 0, stream>>>(tgt, emb, x, xb);

    for (int l = 0; l < LNUM; ++l) {
        const size_t wdd = (size_t)l * DMOD * DMOD;
        const size_t wdf = (size_t)l * DMOD * DFFN;
        convw(Wq + wdd, cwq, DMOD, DMOD);
        convw(Wk + wdd, cwk, DMOD, DMOD);
        convw(Wv + wdd, cwv, DMOD, DMOD);
        convw(Wo + wdd, cwo, DMOD, DMOD);
        convw(Wf1 + wdf, cw1, DMOD, DFFN);
        convw(Wf2 + wdf, cw2, DFFN, DMOD);

        // QKV: bf16 outputs only (consumed by attention WMMAs)
        gemm(xb, cwq, bq + (size_t)l * DMOD, nullptr, nullptr, q16, DMOD, DMOD, 0);
        gemm(xb, cwk, bk + (size_t)l * DMOD, nullptr, nullptr, k16, DMOD, DMOD, 0);
        gemm(xb, cwv, bv + (size_t)l * DMOD, nullptr, nullptr, v16, DMOD, DMOD, 0);

        attention_kernel<<<dim3(SLEN / 16, HEADS, BATCH), 32, 0, stream>>>(
            q16, k16, v16, tgt, ao16);

        // O-projection + residual(x) -> t ; then LN -> x (f32 + bf16)
        gemm(ao16, cwo, bo + (size_t)l * DMOD, x, t, nullptr, DMOD, DMOD, 2);
        layernorm_kernel<<<NTOK, 256, 0, stream>>>(t, g1 + (size_t)l * DMOD,
                                                   be1 + (size_t)l * DMOD, x, xb);

        // FFN: ReLU hidden kept bf16-only; FFN2 adds residual(x)
        gemm(xb, cw1, bf1 + (size_t)l * DFFN, nullptr, nullptr, h16, DFFN, DMOD, 1);
        gemm(h16, cw2, bf2 + (size_t)l * DMOD, x, t, nullptr, DMOD, DFFN, 2);
        layernorm_kernel<<<NTOK, 256, 0, stream>>>(t, g3 + (size_t)l * DMOD,
                                                   be3 + (size_t)l * DMOD, x, xb);
    }

    // final LN + vocab projection
    layernorm_kernel<<<NTOK, 256, 0, stream>>>(x, gN, beN, t, tb);
    convw(Wout, cwT, DMOD, VSZ);
    gemm(tb, cwT, bout, nullptr, out, nullptr, VSZ, DMOD, 0);
}
